// S5Cell_71468255806050
// MI455X (gfx1250) — compile-verified, matrix operand order checked by reading, not run on hardware
//
#include <hip/hip_runtime.h>

typedef __attribute__((ext_vector_type(16))) __bf16 v16bf;
typedef __attribute__((ext_vector_type(8)))  float  v8f;
typedef __attribute__((ext_vector_type(4)))  unsigned v4u;
typedef __attribute__((ext_vector_type(8)))  unsigned v8u;

#define Bsz 16
#define Tsz 8192
#define Msz 128
#define Psz 128
#define Nsz 128
#define CT  128          // timesteps per chunk
#define NC  (Tsz / CT)   // 64 chunks per batch row
#define SU  132          // bf16 LDS row stride (elements)
#define SN  132          // f32  LDS row stride (elements)

// LDS layout (bytes), identical in both passes:
//   0      : uS   f32  [128][132] = 67584   (pass2: reused for crS/ciS after GEMM1)
//   67584  : bdrS bf16 [128][132] = 33792
//   101376 : bdiS bf16 [128][132] = 33792   (contiguous after bdrS -> one TDM tile)
//   135168 : bur  f32  [128][132] = 67584
//   202752 : bui  f32  [128][132] = 67584   -> total 270336
#define LDS_U    0
#define LDS_BDR  67584
#define LDS_BDI  101376
#define LDS_BUR  135168
#define LDS_BUI  202752
#define LDS_TOTAL 270336

// ---------- helpers ----------

__device__ inline unsigned pack2(float a, float b) {
  union { __bf16 h[2]; unsigned u; } c;
  c.h[0] = (__bf16)a; c.h[1] = (__bf16)b;
  return c.u;
}

struct Frag { union { v16bf v; unsigned u[8]; }; };

// 16x32 bf16 WMMA A/B fragment from an LDS [row][k] bf16 array.
__device__ inline void ld_frag(Frag& f, const __bf16* base, int row0, int stride,
                               int kb, int lane) {
  int r = lane & 15, half = lane >> 4;
  const unsigned* p = (const unsigned*)(base + (row0 + r) * stride);
  int k0 = (kb << 5) + (half << 3);
#pragma unroll
  for (int j = 0; j < 4; ++j) {
    f.u[j]     = p[(k0 >> 1) + j];
    f.u[j + 4] = p[((k0 + 16) >> 1) + j];
  }
}

// Same fragment but source is f32 in LDS: convert to bf16 while packing.
__device__ inline void ld_frag_f32(Frag& f, const float* base, int row0, int stride,
                                   int kb, int lane) {
  int r = lane & 15, half = lane >> 4;
  const float* p = base + (row0 + r) * stride;
  int k0 = (kb << 5) + (half << 3);
#pragma unroll
  for (int j = 0; j < 4; ++j) {
    f.u[j]     = pack2(p[k0 + 2 * j],      p[k0 + 2 * j + 1]);
    f.u[j + 4] = pack2(p[k0 + 16 + 2 * j], p[k0 + 17 + 2 * j]);
  }
}

__device__ inline v8f wmma_bf16(const Frag& a, const Frag& b, v8f c) {
  return __builtin_amdgcn_wmma_f32_16x16x32_bf16(false, a.v, false, b.v,
                                                 (short)0, c, false, false);
}

// Issue one TDM 2D tile load global->LDS (D# groups 0/1 per CDNA5 ISA 8.3/8.4).
__device__ inline void tdm_load_2d(unsigned lds_addr, unsigned long long gaddr,
                                   unsigned dsize_log,           // 0:1B 1:2B 2:4B
                                   unsigned tile0, unsigned tile1,
                                   unsigned tdim0, unsigned tdim1,
                                   unsigned long long stride0,
                                   unsigned pad_int, unsigned pad_amt) {
  v4u g0;
  g0[0] = 1u;                                               // count=1 (valid), user mode
  g0[1] = lds_addr;                                         // LDS byte address
  g0[2] = (unsigned)gaddr;                                  // global_addr[31:0]
  g0[3] = ((unsigned)(gaddr >> 32) & 0x01ffffffu) | (2u << 30); // [56:32] | type=2
  v8u g1;
  g1[0] = (dsize_log << 16) | (1u << 20) | (pad_int << 22) | (pad_amt << 25);
  g1[1] = (tdim0 & 0xffffu) << 16;                          // tensor_dim0[15:0]
  g1[2] = (tdim0 >> 16) | ((tdim1 & 0xffffu) << 16);        // dim0 hi | dim1 lo
  g1[3] = (tdim1 >> 16) | (tile0 << 16);                    // dim1 hi | tile_dim0
  g1[4] = tile1 & 0xffffu;                                  // tile_dim1 (tile_dim2=0)
  g1[5] = (unsigned)stride0;                                // tensor_dim0_stride lo
  g1[6] = (unsigned)(stride0 >> 32) & 0xffffu;              // stride hi (dim1_stride=0)
  g1[7] = 0u;
  asm volatile("tensor_load_to_lds %0, %1" :: "s"(g0), "s"(g1) : "memory");
}

__device__ inline unsigned lds_off(const void* p) {
  return (unsigned)(uintptr_t)p;   // LDS aperture: addr[31:0] is the LDS byte address
}

// GEMM1: Bu[t,n] (complex f32) = u(t,:) x Bd^T via bf16 WMMA, u read as f32.
// Software-pipelined: next B-fragment pair loads into fresh registers while the
// current pair feeds the WMMA pipe -> partial s_wait_dscnt instead of full drain.
__device__ inline void gemm_proj(const float* uS, const __bf16* bdrS, const __bf16* bdiS,
                                 float* bur, float* bui, int tid) {
  int lane = tid & 31, w = tid >> 5;
  int tr = w << 4;                       // 16-timestep band per wave
  Frag a[4];
#pragma unroll
  for (int kb = 0; kb < 4; ++kb) ld_frag_f32(a[kb], uS, tr, SN, kb, lane);
  Frag curR, curI, nxtR, nxtI;
  ld_frag(curR, bdrS, 0, SU, 0, lane);
  ld_frag(curI, bdiS, 0, SU, 0, lane);
  v8f zero = {};
  v8f ar = zero, ai = zero;
#pragma unroll
  for (int idx = 0; idx < 32; ++idx) {
    int nt = idx >> 2, kb = idx & 3;
    if (idx < 31) {
      int nn = (idx + 1) >> 2, nk = (idx + 1) & 3;
      ld_frag(nxtR, bdrS, nn << 4, SU, nk, lane);
      ld_frag(nxtI, bdiS, nn << 4, SU, nk, lane);
    }
    ar = wmma_bf16(a[kb], curR, ar);
    ai = wmma_bf16(a[kb], curI, ai);
    if (kb == 3) {
      int n  = (nt << 4) + (lane & 15);
      int rh = (lane >> 4) << 3;
#pragma unroll
      for (int r = 0; r < 8; ++r) {
        bur[(tr + rh + r) * SN + n] = ar[r];
        bui[(tr + rh + r) * SN + n] = ai[r];
      }
      ar = zero; ai = zero;
    }
    if (idx < 31) { curR = nxtR; curI = nxtI; }
  }
}

// ---------- kernels ----------

// Precompute ZOH-discretized operators: Ld, Ld^CT, Bd (bf16), C (bf16, -C_im).
__global__ void s5_setup(const float* __restrict__ lam, const float* __restrict__ Bp,
                         const float* __restrict__ Cp, const float* __restrict__ delta,
                         __bf16* __restrict__ bd, __bf16* __restrict__ cc,
                         float2* __restrict__ Ld, float2* __restrict__ LdL) {
  int n = threadIdx.x;
  float d  = delta[0];
  float lr = -expf(lam[n]);
  float li = lam[Nsz + n];
  float ea  = expf(lr * d);
  float ldr = ea * cosf(li * d);
  float ldi = ea * sinf(li * d);
  Ld[n] = make_float2(ldr, ldi);
  float ar = ldr, ai = ldi;                 // Ld^128 via 7 complex squarings
  for (int i = 0; i < 7; ++i) { float t = ar * ar - ai * ai; ai = 2.f * ar * ai; ar = t; }
  LdL[n] = make_float2(ar, ai);
  float nr = ldr - 1.f, ni = ldi;           // w = (Ld-1)/Lambda
  float den = lr * lr + li * li;
  float wr = (nr * lr + ni * li) / den;
  float wi = (ni * lr - nr * li) / den;
  for (int m = 0; m < Msz; ++m) {
    float br = Bp[n * Msz + m], bi = Bp[Nsz * Msz + n * Msz + m];
    bd[n * Msz + m]             = (__bf16)(wr * br - wi * bi);
    bd[Nsz * Msz + n * Msz + m] = (__bf16)(wr * bi + wi * br);
  }
  for (int k = 0; k < Nsz; ++k) {           // store C_re and -C_im
    cc[n * Nsz + k]             = (__bf16)Cp[n * Nsz + k];
    cc[Psz * Nsz + n * Nsz + k] = (__bf16)(-Cp[Psz * Nsz + n * Nsz + k]);
  }
}

// Pass 1: TDM-stage u+Bd, GEMM1, chunk-local scan (x0=0) -> final state only.
__global__ void __launch_bounds__(256)
s5_pass1(const float* __restrict__ u, const __bf16* __restrict__ bd,
         const float2* __restrict__ Ld, float2* __restrict__ last) {
  extern __shared__ char smem[];
  float*  uS   = (float*)(smem + LDS_U);
  __bf16* bdrS = (__bf16*)(smem + LDS_BDR);
  __bf16* bdiS = (__bf16*)(smem + LDS_BDI);
  float*  bur  = (float*)(smem + LDS_BUR);
  float*  bui  = (float*)(smem + LDS_BUI);
  int tid = threadIdx.x;
  int b = blockIdx.x / NC, c = blockIdx.x % NC;
  const float* ug = u + ((size_t)(b * Tsz + c * CT)) * Msz;

  if (tid < 32) {
    // u: 128x128 f32 tile, LDS pad 4 dwords / 128 dwords -> stride 132 f32
    tdm_load_2d(lds_off(uS), (unsigned long long)(uintptr_t)ug, 2u,
                Msz, CT, Msz, CT, Msz, 6u, 3u);
    // Bd re||im: 256x128 bf16 tile, pad 2 dwords / 64 dwords -> stride 132 bf16
    tdm_load_2d(lds_off(bdrS), (unsigned long long)(uintptr_t)bd, 1u,
                Msz, 2 * Nsz, Msz, 2 * Nsz, Msz, 5u, 1u);
    __builtin_amdgcn_s_wait_tensorcnt(0);
  }
  __syncthreads();
  gemm_proj(uS, bdrS, bdiS, bur, bui, tid);
  __syncthreads();
  if (tid < Nsz) {
    float2 ld = Ld[tid];
    float xr = 0.f, xi = 0.f;
    for (int t = 0; t < CT; ++t) {
      float br = bur[t * SN + tid], bi = bui[t * SN + tid];
      float nr = ld.x * xr - ld.y * xi + br;
      float ni = ld.x * xi + ld.y * xr + bi;
      xr = nr; xi = ni;
    }
    last[(b * NC + c) * Nsz + tid] = make_float2(xr, xi);
  }
}

// Carry scan across chunks: s_{c+1} = Ld^CT * s_c + last_c ; carry_c = s_c.
__global__ void s5_carry(const float2* __restrict__ last, const float2* __restrict__ LdL,
                         float2* __restrict__ carry) {
  int b = blockIdx.x, n = threadIdx.x;
  float2 a = LdL[n];
  float sr = 0.f, si = 0.f;
  for (int c = 0; c < NC; ++c) {
    size_t i = (size_t)(b * NC + c) * Nsz + n;
    carry[i] = make_float2(sr, si);
    float2 l = last[i];
    float nr = a.x * sr - a.y * si + l.x;
    float ni = a.x * si + a.y * sr + l.y;
    sr = nr; si = ni;
  }
}

// Pass 2: GEMM1 again, scan seeded with carry -> global x, GEMM2 -> y.
// C tiles are TDM-DMA'd into the dead u region concurrently with the scan.
__global__ void __launch_bounds__(256)
s5_pass2(const float* __restrict__ u, const __bf16* __restrict__ bd,
         const __bf16* __restrict__ cc, const float2* __restrict__ Ld,
         const float2* __restrict__ carry, float* __restrict__ y) {
  extern __shared__ char smem[];
  float*  uS   = (float*)(smem + LDS_U);
  __bf16* bdrS = (__bf16*)(smem + LDS_BDR);
  __bf16* bdiS = (__bf16*)(smem + LDS_BDI);
  float*  bur  = (float*)(smem + LDS_BUR);
  float*  bui  = (float*)(smem + LDS_BUI);
  __bf16* crS  = (__bf16*)(smem + LDS_U);          // reuse u region after GEMM1
  __bf16* ciS  = (__bf16*)(smem + LDS_U + 33792);
  int tid = threadIdx.x;
  int b = blockIdx.x / NC, c = blockIdx.x % NC;
  const float* ug = u + ((size_t)(b * Tsz + c * CT)) * Msz;

  if (tid < 32) {
    tdm_load_2d(lds_off(uS), (unsigned long long)(uintptr_t)ug, 2u,
                Msz, CT, Msz, CT, Msz, 6u, 3u);
    tdm_load_2d(lds_off(bdrS), (unsigned long long)(uintptr_t)bd, 1u,
                Msz, 2 * Nsz, Msz, 2 * Nsz, Msz, 5u, 1u);
    __builtin_amdgcn_s_wait_tensorcnt(0);
  }
  __syncthreads();
  gemm_proj(uS, bdrS, bdiS, bur, bui, tid);
  __syncthreads();
  if (tid < 32) {
    // C re||-im into the dead u region; overlaps with the scan below
    tdm_load_2d(lds_off(crS), (unsigned long long)(uintptr_t)cc, 1u,
                Nsz, 2 * Psz, Nsz, 2 * Psz, Nsz, 5u, 1u);
  }
  if (tid < Nsz) {                       // scan with carry seed -> global x in LDS
    float2 ld = Ld[tid];
    float2 cv = carry[(b * NC + c) * Nsz + tid];
    float xr = cv.x, xi = cv.y;
    for (int t = 0; t < CT; ++t) {
      float br = bur[t * SN + tid], bi = bui[t * SN + tid];
      float nr = ld.x * xr - ld.y * xi + br;
      float ni = ld.x * xi + ld.y * xr + bi;
      xr = nr; xi = ni;
      bur[t * SN + tid] = xr;
      bui[t * SN + tid] = xi;
    }
  }
  if (tid < 32) __builtin_amdgcn_s_wait_tensorcnt(0);
  __syncthreads();
  // GEMM2: y[t,p] = 2*(x_re @ C_re^T + x_im @ (-C_im)^T), pipelined like GEMM1.
  int lane = tid & 31, w = tid >> 5;
  int tr = w << 4;
  Frag xrF[4], xiF[4];
#pragma unroll
  for (int kb = 0; kb < 4; ++kb) {
    ld_frag_f32(xrF[kb], bur, tr, SN, kb, lane);
    ld_frag_f32(xiF[kb], bui, tr, SN, kb, lane);
  }
  size_t ybase = ((size_t)(b * Tsz + c * CT)) * Psz;
  Frag curR, curI, nxtR, nxtI;
  ld_frag(curR, crS, 0, SU, 0, lane);
  ld_frag(curI, ciS, 0, SU, 0, lane);
  v8f zero = {};
  v8f acc = zero;
#pragma unroll
  for (int idx = 0; idx < 32; ++idx) {
    int pt = idx >> 2, kb = idx & 3;
    if (idx < 31) {
      int np = (idx + 1) >> 2, nk = (idx + 1) & 3;
      ld_frag(nxtR, crS, np << 4, SU, nk, lane);
      ld_frag(nxtI, ciS, np << 4, SU, nk, lane);
    }
    acc = wmma_bf16(xrF[kb], curR, acc);
    acc = wmma_bf16(xiF[kb], curI, acc);
    if (kb == 3) {
      int p  = (pt << 4) + (lane & 15);
      int rh = (lane >> 4) << 3;
#pragma unroll
      for (int r = 0; r < 8; ++r)
        y[ybase + (size_t)(tr + rh + r) * Psz + p] = 2.0f * acc[r];
      acc = zero;
    }
    if (idx < 31) { curR = nxtR; curI = nxtI; }
  }
}

// ---------- launcher ----------

extern "C" void kernel_launch(void* const* d_in, const int* in_sizes, int n_in,
                              void* d_out, int out_size, void* d_ws, size_t ws_size,
                              hipStream_t stream) {
  (void)in_sizes; (void)n_in; (void)out_size; (void)ws_size;
  const float* u     = (const float*)d_in[0];
  const float* lam   = (const float*)d_in[1];
  const float* Bp    = (const float*)d_in[2];
  const float* Cp    = (const float*)d_in[3];
  const float* delta = (const float*)d_in[4];
  float* y = (float*)d_out;

  char* ws = (char*)d_ws;
  __bf16* bd    = (__bf16*)(ws);                       // 2*128*128 bf16 = 64 KB
  __bf16* cc    = (__bf16*)(ws + 65536);               // 2*128*128 bf16 = 64 KB
  float2* Ld    = (float2*)(ws + 131072);              // 1 KB
  float2* LdL   = (float2*)(ws + 132096);              // 1 KB
  float2* last  = (float2*)(ws + 133120);              // 16*64*128 float2 = 1 MB
  float2* carry = (float2*)(ws + 133120 + (size_t)Bsz * NC * Nsz * sizeof(float2));

  s5_setup<<<1, 128, 0, stream>>>(lam, Bp, Cp, delta, bd, cc, Ld, LdL);
  s5_pass1<<<Bsz * NC, 256, LDS_TOTAL, stream>>>(u, bd, Ld, last);
  s5_carry<<<Bsz, 128, 0, stream>>>(last, LdL, carry);
  s5_pass2<<<Bsz * NC, 256, LDS_TOTAL, stream>>>(u, bd, cc, Ld, carry, y);
}
